// FDFA_40415642255453
// MI455X (gfx1250) — compile-verified, matrix operand order checked by reading, not run on hardware
//
#include <hip/hip_runtime.h>
#include <hip/hip_bf16.h>

// ---------------------------------------------------------------------------
// CDNA5 / gfx1250 implementation. wave32, WMMA bf16 16x16x32, f32 accumulate.
// ---------------------------------------------------------------------------

typedef __attribute__((ext_vector_type(16))) __bf16 v16bf;
typedef __attribute__((ext_vector_type(8)))  __bf16 v8bf;
typedef __attribute__((ext_vector_type(8)))  float  v8f;

static constexpr int  B_   = 4;
static constexpr int  C_   = 96;
static constexpr int  H_   = 256;
static constexpr int  W_   = 256;
static constexpr int  HEAD = 8;
static constexpr int  CPH  = 12;                  // channels per head
static constexpr long HW   = (long)H_ * W_;       // 65536
static constexpr long CHW  = (long)C_ * HW;
static constexpr int  T_   = 256;                 // tokens per (b,head)
static constexpr int  D_   = W_ * CPH;            // 3072 token dim
static constexpr long NELEM = (long)B_ * CHW;     // 25,165,824
static constexpr int  BN   = B_ * HEAD;           // 32

#define DEV static __device__ __forceinline__

DEV v8f vzero8() { v8f z = {0.f,0.f,0.f,0.f,0.f,0.f,0.f,0.f}; return z; }

// Fragment load from a row-major LDS matrix [rows][ld] (ld multiple of 8).
// Lane mapping per CDNA5 ISA 16-bit A layout (also used for B where the LDS
// tile is stored transposed so lanes index the N dimension):
//   lane<16 : row = row0+lane,     k = k0 + {0..7, 16..23}
//   lane>=16: row = row0+lane-16,  k = k0 + {8..15, 24..31}
// Each lane = two contiguous 16B chunks -> two ds_load_b128.
DEV v16bf frag_ld(const __bf16* sm, int row0, int k0, int ld) {
  int lane = threadIdx.x & 31;
  const __bf16* p = sm + (long)(row0 + (lane & 15)) * ld + k0 + ((lane >> 4) << 3);
  v8bf lo = *(const v8bf*)p;
  v8bf hi = *(const v8bf*)(p + 16);
  return __builtin_shufflevector(lo, hi, 0,1,2,3,4,5,6,7,8,9,10,11,12,13,14,15);
}

DEV v8f wmma_bf16(v16bf a, v16bf b, v8f c) {
  // D = A x B + C ; f32 accumulate, no modifiers
  return __builtin_amdgcn_wmma_f32_16x16x32_bf16(false, a, false, b, (short)0, c,
                                                 false, false);
}

// C/D 16x16 f32 tile store: VGPR r -> M = r (lanes 0-15) / r+8 (lanes 16-31),
// N = lane & 15.
DEV void store_tile_lds(float* sm, int ld, int m0, int n0, v8f c) {
  int lane = threadIdx.x & 31;
  int m = m0 + ((lane >> 4) << 3);
  int n = n0 + (lane & 15);
  #pragma unroll
  for (int r = 0; r < 8; ++r) sm[(long)(m + r) * ld + n] = c[r];
}

// ---------------------------------------------------------------------------
// 1) Channel LayerNorm (NCHW, over C). One thread per pixel, coalesced per ch.
// ---------------------------------------------------------------------------
__global__ __launch_bounds__(256) void ln_kernel(const float* __restrict__ x,
                                                 const float* __restrict__ w,
                                                 const float* __restrict__ b,
                                                 float* __restrict__ y) {
  long pix = (long)blockIdx.x * 256 + threadIdx.x;   // < B*HW
  long bb = pix / HW, sp = pix - bb * HW;
  const float* xp = x + bb * CHW + sp;
  float s = 0.f, ss = 0.f;
  #pragma unroll 16
  for (int c = 0; c < C_; ++c) { float v = xp[(long)c * HW]; s += v; ss += v * v; }
  float mu = s * (1.f / C_);
  float var = ss * (1.f / C_) - mu * mu;
  float rs = rsqrtf(var + 1e-5f);
  float* yp = y + bb * CHW + sp;
  #pragma unroll 16
  for (int c = 0; c < C_; ++c)
    yp[(long)c * HW] = (xp[(long)c * HW] - mu) * rs * w[c] + b[c];
}

// ---------------------------------------------------------------------------
// 2) depthwise 1x7 + 1x11 conv along W, summed. One block per (b, ch, row).
// ---------------------------------------------------------------------------
__global__ __launch_bounds__(256) void dw_kernel(const float* __restrict__ x,
                                                 const float* __restrict__ w7,
                                                 const float* __restrict__ b7,
                                                 const float* __restrict__ w11,
                                                 const float* __restrict__ b11,
                                                 float* __restrict__ y) {
  int row = blockIdx.x, ch = blockIdx.y, bb = blockIdx.z;
  __shared__ float line[W_ + 10];
  int t = threadIdx.x;
  const float* xp = x + ((long)bb * C_ + ch) * HW + (long)row * W_;
  line[5 + t] = xp[t];
  if (t < 5) { line[t] = 0.f; line[5 + W_ + t] = 0.f; }
  __syncthreads();
  float a7 = b7[ch];
  #pragma unroll
  for (int k = 0; k < 7; ++k) a7 += w7[ch * 7 + k] * line[5 + t - 3 + k];
  float a11 = b11[ch];
  #pragma unroll
  for (int k = 0; k < 11; ++k) a11 += w11[ch * 11 + k] * line[5 + t - 5 + k];
  y[((long)bb * C_ + ch) * HW + (long)row * W_ + t] = a7 + a11;
}

// ---------------------------------------------------------------------------
// 3) pack proj_w -> bf16
// ---------------------------------------------------------------------------
__global__ void packw_kernel(const float* __restrict__ w, __bf16* __restrict__ wb) {
  int i = blockIdx.x * 256 + threadIdx.x;
  if (i < C_ * C_) wb[i] = (__bf16)w[i];
}

// ---------------------------------------------------------------------------
// 4) 1x1 conv (y = W x + s*bias [+ r1 + r2]) as WMMA GEMM.
//    Block: 256 pixels x full M=96, K=96. 8 waves; each wave owns two
//    adjacent 16-pixel N-tiles (A-fragment reused across both).
//    X tile stored transposed [n][k] in LDS -> contiguous b128 fragment loads.
//    Templated on residual presence -> branch-free epilogue.
// ---------------------------------------------------------------------------
static constexpr int PC_NT  = 256;
static constexpr int PC_LDW = C_ + 8;   // 104 (row stride 208B, bank-spread)
static constexpr int PC_LDB = C_ + 8;   // 104
static constexpr int PCONV_SMEM = C_ * PC_LDW * 2 + PC_NT * PC_LDB * 2; // 73216

template <bool HAS_RES>
__global__ __launch_bounds__(256) void pconv_kernel(const float* __restrict__ x,
                                                    const __bf16* __restrict__ wb,
                                                    const float* __restrict__ bias,
                                                    float bscale,
                                                    const float* __restrict__ r1,
                                                    const float* __restrict__ r2,
                                                    float* __restrict__ y) {
  extern __shared__ __align__(16) char psm[];
  __bf16* sW = (__bf16*)psm;                       // [o][PC_LDW]
  __bf16* sX = (__bf16*)(psm + C_ * PC_LDW * 2);   // [n][PC_LDB]
  int t = threadIdx.x;
  long p0 = (long)blockIdx.x * PC_NT;
  long bb = p0 / HW, sp0 = p0 - bb * HW;
  const float* xp = x + bb * CHW + sp0;
  #pragma unroll
  for (int i = 0; i < (C_ * C_) / 256; ++i) {      // 36
    int idx = t + i * 256;
    int o = idx / C_, c = idx % C_;
    sW[o * PC_LDW + c] = wb[idx];
  }
  #pragma unroll
  for (int i = 0; i < (C_ * PC_NT) / 256; ++i) {   // 96, coalesced f32 reads
    int idx = t + i * 256;
    int k = idx >> 8, n = idx & 255;
    sX[n * PC_LDB + k] = (__bf16)xp[(long)k * HW + n];
  }
  __syncthreads();
  int wv = t >> 5, lane = t & 31;
  int n0 = wv * 32;
  #pragma unroll
  for (int mt = 0; mt < 6; ++mt) {
    v8f a0 = vzero8(), a1 = vzero8();
    // split loads from math: no VALU touching live WMMA A/B regs between WMMAs
    v16bf af[3], b0[3], b1[3];
    #pragma unroll
    for (int kt = 0; kt < 3; ++kt) {
      af[kt] = frag_ld(sW, mt * 16, kt * 32, PC_LDW);
      b0[kt] = frag_ld(sX, n0,      kt * 32, PC_LDB);
      b1[kt] = frag_ld(sX, n0 + 16, kt * 32, PC_LDB);
    }
    #pragma unroll
    for (int kt = 0; kt < 3; ++kt) {
      a0 = wmma_bf16(af[kt], b0[kt], a0);
      a1 = wmma_bf16(af[kt], b1[kt], a1);
    }
    int m = mt * 16 + ((lane >> 4) << 3);
    int n = n0 + (lane & 15);
    long gp = bb * CHW + sp0 + n;
    float r0v[8], r1v[8];
    if constexpr (HAS_RES) {
      #pragma unroll
      for (int r = 0; r < 8; ++r) {
        long gi = gp + (long)(m + r) * HW;
        r0v[r] = r1[gi] + r2[gi];
        r1v[r] = r1[gi + 16] + r2[gi + 16];
      }
    }
    #pragma unroll
    for (int r = 0; r < 8; ++r) {
      long gi = gp + (long)(m + r) * HW;
      float bv = bscale * bias[m + r];
      float o0v = a0[r] + bv;
      float o1v = a1[r] + bv;
      if constexpr (HAS_RES) { o0v += r0v[r]; o1v += r1v[r]; }
      y[gi] = o0v;
      y[gi + 16] = o1v;
    }
  }
}

// ---------------------------------------------------------------------------
// 5) token row inverse L2 norms. mode 0: tok_h (token = H row), mode 1: tok_w.
// ---------------------------------------------------------------------------
__global__ __launch_bounds__(256) void rownorm_kernel(const float* __restrict__ x,
                                                      int mode,
                                                      float* __restrict__ inv) {
  int i = blockIdx.x, bn = blockIdx.y;
  int bb = bn >> 3, n = bn & 7;
  int t = threadIdx.x;
  const float* base = x + (long)bb * CHW + (long)n * CPH * HW;
  float s = 0.f;
  #pragma unroll
  for (int cc = 0; cc < CPH; ++cc) {
    float v = (mode == 0) ? base[(long)cc * HW + (long)i * W_ + t]
                          : base[(long)cc * HW + (long)t * W_ + i];
    s += v * v;
  }
  __shared__ float red[256];
  red[t] = s;
  __syncthreads();
  for (int off = 128; off > 0; off >>= 1) {
    if (t < off) red[t] += red[t + off];
    __syncthreads();
  }
  if (t == 0) inv[bn * T_ + i] = 1.f / fmaxf(sqrtf(red[0]), 1e-12f);
}

// ---------------------------------------------------------------------------
// 6a) tok_h pack: out[bn][i][ww*12+cc] = x[b][n*12+cc][i][ww] * inv, bf16.
// ---------------------------------------------------------------------------
__global__ __launch_bounds__(256) void tokh_kernel(const float* __restrict__ x,
                                                   const float* __restrict__ inv,
                                                   __bf16* __restrict__ out) {
  int i = blockIdx.x, bn = blockIdx.y;
  int bb = bn >> 3, n = bn & 7;
  int t = threadIdx.x;                   // = ww
  float sc = inv ? inv[bn * T_ + i] : 1.f;
  const float* base = x + (long)bb * CHW + (long)n * CPH * HW + (long)i * W_ + t;
  __bf16* op = out + ((long)bn * T_ + i) * D_ + t * CPH;
  #pragma unroll
  for (int cc = 0; cc < CPH; ++cc) op[cc] = (__bf16)(base[(long)cc * HW] * sc);
}

// ---------------------------------------------------------------------------
// 6b) tok_w pack via 16x16 LDS transpose tile (keeps reads+writes coalesced):
//     out[bn][i][hh*12+cc] = x[b][n*12+cc][hh][i] * inv
// ---------------------------------------------------------------------------
__global__ __launch_bounds__(256) void tokw_kernel(const float* __restrict__ x,
                                                   const float* __restrict__ inv,
                                                   __bf16* __restrict__ out) {
  int it = blockIdx.x, ht = blockIdx.y, bn = blockIdx.z;
  int bb = bn >> 3, n = bn & 7;
  int t = threadIdx.x;
  int tx = t & 15, ty = t >> 4;
  int i0 = it * 16, h0 = ht * 16;
  __shared__ float tile[CPH][16][17];
  const float* base = x + (long)bb * CHW + (long)n * CPH * HW;
  #pragma unroll
  for (int cc = 0; cc < CPH; ++cc)
    tile[cc][ty][tx] = base[(long)cc * HW + (long)(h0 + ty) * W_ + i0 + tx];
  __syncthreads();
  int cc = t % CPH, hh = t / CPH;        // valid for t < 192
  for (int ii = 0; ii < 16; ++ii) {
    if (t < 192) {
      float sc = inv ? inv[bn * T_ + i0 + ii] : 1.f;
      out[((long)bn * T_ + i0 + ii) * D_ + (h0 + hh) * CPH + cc] =
          (__bf16)(tile[cc][hh][ii] * sc);
    }
  }
}

// ---------------------------------------------------------------------------
// 7) attention: per (bn, 64-row block). S=QK^T (96 k-chunks of 32, register-
//    staged software pipeline, double-buffered LDS statically unrolled x2,
//    ONE barrier per chunk), register-vectorized softmax, O = P V + Q.
//    LDS: phase-1 double buffers are aliased under the 64KB f32 S scratch.
// ---------------------------------------------------------------------------
static constexpr int LQ = 40, LK = 40, LP = 264, LV = 264;   // padded strides
static constexpr int QBUF = 64 * LQ * 2;       //  5120 B per buffer
static constexpr int KBUF = 256 * LK * 2;      // 20480 B per buffer
static constexpr int PH1  = QBUF + KBUF;       // 25600 B per (Q,K) buffer pair
static constexpr int SSZ  = 64 * 256 * 4;      // 65536 B (covers 2*PH1)
static constexpr int ATTN_SMEM = SSZ + 64 * LP * 2 + 64 * LV * 2 + 256 * 4; // 134144

__global__ __launch_bounds__(256) void attn_kernel(const __bf16* __restrict__ Q,
                                                   const __bf16* __restrict__ K,
                                                   const __bf16* __restrict__ V,
                                                   float* __restrict__ O) {
  extern __shared__ __align__(16) char smem[];
  __bf16* sQ0 = (__bf16*)smem;
  __bf16* sK0 = (__bf16*)(smem + QBUF);
  __bf16* sQ1 = (__bf16*)(smem + PH1);
  __bf16* sK1 = (__bf16*)(smem + PH1 + QBUF);
  float*  sS  = (float*)smem;                         // aliases phase-1 buffers
  __bf16* sP  = (__bf16*)(smem + SSZ);                // 33792 B
  __bf16* sV  = (__bf16*)(smem + SSZ + 64 * LP * 2);  // 33792 B ([n][k])
  float* sRed = (float*)(smem + SSZ + 2 * 64 * LP * 2);

  const int bn = blockIdx.y;
  const int row0 = blockIdx.x * 64;
  const int t = threadIdx.x, lane = t & 31, wv = t >> 5;
  const long qbase = ((long)bn * T_ + row0) * D_;
  const long kbase = (long)bn * T_ * D_;
  const int mt = wv & 3, ng = wv >> 2;         // wave -> (m tile, n group)

  // ---- phase 1: S = Q K^T, 64x256 accumulated in registers -------------
  v8f acc[8];
  #pragma unroll
  for (int j = 0; j < 8; ++j) acc[j] = vzero8();

  const int qr = t >> 2, qc8 = (t & 3) * 8;
  const __bf16* Qg = Q + qbase + (long)qr * D_ + qc8;
  const __bf16* Kg = K + kbase + (long)t * D_;
  v8bf qreg;
  v8bf kreg[4];
  auto fetchQK = [&](int kc) {                 // global -> registers
    qreg = *(const v8bf*)(Qg + kc * 32);
    const __bf16* src = Kg + kc * 32;
    #pragma unroll
    for (int u = 0; u < 4; ++u) kreg[u] = *(const v8bf*)(src + u * 8);
  };
  auto commitQK = [&](__bf16* q, __bf16* k) {  // registers -> LDS
    *(v8bf*)(q + qr * LQ + qc8) = qreg;
    __bf16* dst = k + t * LK;
    #pragma unroll
    for (int u = 0; u < 4; ++u) *(v8bf*)(dst + u * 8) = kreg[u];
  };
  // Load all 8 B-fragments first, then issue 8 back-to-back WMMAs: no VALU
  // writes to live A/B registers between WMMAs -> no WAR hazard NOPs.
  auto computeS = [&](const __bf16* q, const __bf16* k) {
    v16bf a = frag_ld(q, mt * 16, 0, LQ);
    v16bf bfr[8];
    #pragma unroll
    for (int j = 0; j < 8; ++j) bfr[j] = frag_ld(k, (ng * 8 + j) * 16, 0, LK);
    #pragma unroll
    for (int j = 0; j < 8; ++j) acc[j] = wmma_bf16(a, bfr[j], acc[j]);
  };

  fetchQK(0);
  commitQK(sQ0, sK0);
  __syncthreads();
  #pragma unroll 1
  for (int kc = 0; kc < D_ / 32; kc += 2) {    // statically double-buffered
    fetchQK(kc + 1);                           // overlap global latency w/ WMMA
    computeS(sQ0, sK0);
    commitQK(sQ1, sK1);
    __syncthreads();
    if (kc + 2 < D_ / 32) fetchQK(kc + 2);
    computeS(sQ1, sK1);
    if (kc + 2 < D_ / 32) commitQK(sQ0, sK0);
    __syncthreads();
  }

  // ---- phase 2: softmax (S scratch aliases the dead phase-1 buffers) ----
  #pragma unroll
  for (int j = 0; j < 8; ++j)
    store_tile_lds(sS, 256, mt * 16, (ng * 8 + j) * 16, acc[j]);
  __syncthreads();
  {
    int row = t >> 2, seg = t & 3;            // 4 threads per row, 64 cols each
    float4* sr4 = (float4*)(sS + row * 256 + seg * 64);
    float4 vv[16];
    #pragma unroll
    for (int c = 0; c < 16; ++c) vv[c] = sr4[c];         // 16x ds_load_b128
    float mx = -3.4e38f;
    #pragma unroll
    for (int c = 0; c < 16; ++c)
      mx = fmaxf(mx, fmaxf(fmaxf(vv[c].x, vv[c].y), fmaxf(vv[c].z, vv[c].w)));
    sRed[t] = mx;
    __syncthreads();
    mx = fmaxf(fmaxf(sRed[row * 4 + 0], sRed[row * 4 + 1]),
               fmaxf(sRed[row * 4 + 2], sRed[row * 4 + 3]));
    float sum = 0.f;
    #pragma unroll
    for (int c = 0; c < 16; ++c) {
      vv[c].x = __expf(vv[c].x - mx);
      vv[c].y = __expf(vv[c].y - mx);
      vv[c].z = __expf(vv[c].z - mx);
      vv[c].w = __expf(vv[c].w - mx);
      sum += (vv[c].x + vv[c].y) + (vv[c].z + vv[c].w);
    }
    __syncthreads();
    sRed[t] = sum;
    __syncthreads();
    sum = sRed[row * 4 + 0] + sRed[row * 4 + 1] + sRed[row * 4 + 2] + sRed[row * 4 + 3];
    float isum = 1.f / sum;
    __bf16* pr = sP + row * LP + seg * 64;
    #pragma unroll
    for (int c = 0; c < 8; ++c) {                        // 8x 16B packed stores
      v8bf pk;
      float4 va = vv[2 * c], vb = vv[2 * c + 1];
      pk[0] = (__bf16)(va.x * isum); pk[1] = (__bf16)(va.y * isum);
      pk[2] = (__bf16)(va.z * isum); pk[3] = (__bf16)(va.w * isum);
      pk[4] = (__bf16)(vb.x * isum); pk[5] = (__bf16)(vb.y * isum);
      pk[6] = (__bf16)(vb.z * isum); pk[7] = (__bf16)(vb.w * isum);
      *(v8bf*)(pr + c * 8) = pk;
    }
  }
  __syncthreads();

  // ---- phase 3: O = P V + Q (register-staged V prefetch) ---------------
  const int nt0 = ng * 2;                      // wave -> two 16-col O tiles
  const __bf16* Vg = V + kbase + (long)t * D_;
  v8bf vreg[8];
  auto fetchV = [&](int dc) {
    const __bf16* src = Vg + dc * 64;
    #pragma unroll
    for (int u = 0; u < 8; ++u) vreg[u] = *(const v8bf*)(src + u * 8);
  };
  auto commitV = [&]() {                       // row t -> column t of sV[n][k]
    #pragma unroll
    for (int u = 0; u < 8; ++u) {
      #pragma unroll
      for (int e = 0; e < 8; ++e) sV[(u * 8 + e) * LV + t] = vreg[u][e];
    }
  };

  fetchV(0);
  commitV();
  __syncthreads();
  #pragma unroll 1
  for (int dc = 0; dc < D_ / 64; ++dc) {       // 48 chunks
    if (dc + 1 < D_ / 64) fetchV(dc + 1);      // global latency hidden by WMMA
    v8f o0 = vzero8(), o1 = vzero8();
    #pragma unroll
    for (int kh = 0; kh < 2; ++kh) {           // loads-then-math half chains
      v16bf af[4], bf0[4], bf1[4];
      #pragma unroll
      for (int kt = 0; kt < 4; ++kt) {
        int kk = kh * 4 + kt;
        af[kt]  = frag_ld(sP, mt * 16,         kk * 32, LP);
        bf0[kt] = frag_ld(sV, nt0 * 16,        kk * 32, LV);
        bf1[kt] = frag_ld(sV, (nt0 + 1) * 16,  kk * 32, LV);
      }
      #pragma unroll
      for (int kt = 0; kt < 4; ++kt) {
        o0 = wmma_bf16(af[kt], bf0[kt], o0);
        o1 = wmma_bf16(af[kt], bf1[kt], o1);
      }
    }
    {
      int m = mt * 16 + ((lane >> 4) << 3);
      int nn = lane & 15;
      #pragma unroll
      for (int r = 0; r < 8; ++r) {
        long gi0 = qbase + (long)(m + r) * D_ + dc * 64 + nt0 * 16 + nn;
        O[gi0] = o0[r] + (float)Q[gi0];
        long gi1 = gi0 + 16;
        O[gi1] = o1[r] + (float)Q[gi1];
      }
    }
    __syncthreads();                            // everyone done reading sV
    if (dc + 1 < D_ / 64) {
      commitV();
      __syncthreads();                          // sV ready for next chunk
    }
  }
}

// ---------------------------------------------------------------------------
// 8) untokenize + sum: s = untok_h(O3) + untok_w(O4), NCHW f32.
//    Block per (hh, bn); thread = ww. Both gathers are 3x contiguous b128
//    loads per thread; 12 fully-coalesced b32 stores.
// ---------------------------------------------------------------------------
__global__ __launch_bounds__(256) void untoksum_kernel(const float* __restrict__ O3,
                                                       const float* __restrict__ O4,
                                                       float* __restrict__ s) {
  int hh = blockIdx.x, bn = blockIdx.y;
  int bb = bn >> 3, n = bn & 7;
  int t = threadIdx.x;                  // = ww
  const float4* r3 = (const float4*)(O3 + ((long)bn * T_ + hh) * D_ + t * CPH);
  const float4* r4 = (const float4*)(O4 + ((long)bn * T_ + t) * D_ + hh * CPH);
  float4 a0 = r3[0], a1 = r3[1], a2 = r3[2];
  float4 b0 = r4[0], b1 = r4[1], b2 = r4[2];
  float* sp = s + (long)bb * CHW + (long)n * CPH * HW + (long)hh * W_ + t;
  sp[0 * HW]  = a0.x + b0.x;
  sp[1 * HW]  = a0.y + b0.y;
  sp[2 * HW]  = a0.z + b0.z;
  sp[3 * HW]  = a0.w + b0.w;
  sp[4 * HW]  = a1.x + b1.x;
  sp[5 * HW]  = a1.y + b1.y;
  sp[6 * HW]  = a1.z + b1.z;
  sp[7 * HW]  = a1.w + b1.w;
  sp[8 * HW]  = a2.x + b2.x;
  sp[9 * HW]  = a2.y + b2.y;
  sp[10 * HW] = a2.z + b2.z;
  sp[11 * HW] = a2.w + b2.w;
}

// ---------------------------------------------------------------------------
extern "C" void kernel_launch(void* const* d_in, const int* in_sizes, int n_in,
                              void* d_out, int out_size, void* d_ws, size_t ws_size,
                              hipStream_t stream) {
  const float* x1   = (const float*)d_in[0];
  const float* x2   = (const float*)d_in[1];
  const float* ln1w = (const float*)d_in[2];
  const float* ln1b = (const float*)d_in[3];
  const float* ln2w = (const float*)d_in[4];
  const float* ln2b = (const float*)d_in[5];
  const float* pw   = (const float*)d_in[6];
  const float* pb   = (const float*)d_in[7];
  const float* c11w = (const float*)d_in[8];
  const float* c11b = (const float*)d_in[9];
  const float* c12w = (const float*)d_in[10];
  const float* c12b = (const float*)d_in[11];
  const float* c21w = (const float*)d_in[12];
  const float* c21b = (const float*)d_in[13];
  const float* c22w = (const float*)d_in[14];
  const float* c22b = (const float*)d_in[15];
  (void)in_sizes; (void)n_in; (void)out_size; (void)ws_size;

  char* wsp = (char*)d_ws;
  auto alloc = [&](size_t bytes) {
    char* p = wsp; wsp += (bytes + 255) & ~size_t(255); return p;
  };
  const size_t F = NELEM * sizeof(float);          // one NCHW f32 tensor
  const size_t Tb = (size_t)BN * T_ * D_ * 2;      // one tokenized bf16 tensor
  float*  ln1 = (float*)alloc(F);
  float*  ln2 = (float*)alloc(F);
  float*  t1  = (float*)alloc(F);                  // later reused as O3
  float*  t2  = (float*)alloc(F);                  // later reused as O4
  float*  p1  = (float*)alloc(F);                  // later reused as ssum
  float*  p2  = (float*)alloc(F);
  __bf16* k1b = (__bf16*)alloc(Tb);
  __bf16* v1b = (__bf16*)alloc(Tb);
  __bf16* k2b = (__bf16*)alloc(Tb);
  __bf16* v2b = (__bf16*)alloc(Tb);
  __bf16* q1b = (__bf16*)alloc(Tb);
  __bf16* q2b = (__bf16*)alloc(Tb);
  __bf16* wbf = (__bf16*)alloc(C_ * C_ * 2);
  float*  ik1 = (float*)alloc(BN * T_ * 4);
  float*  ik2 = (float*)alloc(BN * T_ * 4);
  float*  iq1 = (float*)alloc(BN * T_ * 4);
  float*  iq2 = (float*)alloc(BN * T_ * 4);
  float*  O3   = t1;
  float*  O4   = t2;
  float*  ssum = p1;

  const int PIX_BLOCKS   = (int)((B_ * HW) / 256);       // 1024
  const int PCONV_BLOCKS = (int)((B_ * HW) / PC_NT);     // 1024

  // 1) layernorm
  ln_kernel<<<PIX_BLOCKS, 256, 0, stream>>>(x1, ln1w, ln1b, ln1);
  ln_kernel<<<PIX_BLOCKS, 256, 0, stream>>>(x2, ln2w, ln2b, ln2);

  // 2) depthwise convs
  dw_kernel<<<dim3(H_, C_, B_), 256, 0, stream>>>(ln1, c11w, c11b, c12w, c12b, t1);
  dw_kernel<<<dim3(H_, C_, B_), 256, 0, stream>>>(ln2, c21w, c21b, c22w, c22b, t2);

  // 3) weight pack + mid 1x1 convs (WMMA)
  packw_kernel<<<(C_ * C_ + 255) / 256, 256, 0, stream>>>(pw, wbf);
  pconv_kernel<false><<<PCONV_BLOCKS, 256, PCONV_SMEM, stream>>>(
      t1, wbf, pb, 1.f, nullptr, nullptr, p1);
  pconv_kernel<false><<<PCONV_BLOCKS, 256, PCONV_SMEM, stream>>>(
      t2, wbf, pb, 1.f, nullptr, nullptr, p2);

  // 4) row norms (k1 <- tok_h(ln1), k2 <- tok_w(ln2), q2 <- tok_h(p1), q1 <- tok_w(p2))
  rownorm_kernel<<<dim3(T_, BN), 256, 0, stream>>>(ln1, 0, ik1);
  rownorm_kernel<<<dim3(T_, BN), 256, 0, stream>>>(ln2, 1, ik2);
  rownorm_kernel<<<dim3(T_, BN), 256, 0, stream>>>(p1, 0, iq2);
  rownorm_kernel<<<dim3(T_, BN), 256, 0, stream>>>(p2, 1, iq1);

  // 5) tokenized bf16 packs
  tokh_kernel<<<dim3(T_, BN), 256, 0, stream>>>(ln1, ik1, k1b);
  tokh_kernel<<<dim3(T_, BN), 256, 0, stream>>>(ln1, nullptr, v1b);
  tokh_kernel<<<dim3(T_, BN), 256, 0, stream>>>(p1, iq2, q2b);
  tokw_kernel<<<dim3(16, 16, BN), 256, 0, stream>>>(ln2, ik2, k2b);
  tokw_kernel<<<dim3(16, 16, BN), 256, 0, stream>>>(ln2, nullptr, v2b);
  tokw_kernel<<<dim3(16, 16, BN), 256, 0, stream>>>(p2, iq1, q1b);

  // 6) attentions (WMMA): out3 = softmax(q1 k1^T) v1 + q1 ; out4 likewise
  attn_kernel<<<dim3(T_ / 64, BN), 256, ATTN_SMEM, stream>>>(q1b, k1b, v1b, O3);
  attn_kernel<<<dim3(T_ / 64, BN), 256, ATTN_SMEM, stream>>>(q2b, k2b, v2b, O4);

  // 7) untokenize + sum, then final 1x1 conv with 2x bias and residuals
  untoksum_kernel<<<dim3(H_, BN), 256, 0, stream>>>(O3, O4, ssum);
  pconv_kernel<true><<<PCONV_BLOCKS, 256, PCONV_SMEM, stream>>>(
      ssum, wbf, pb, 2.f, ln1, ln2, (float*)d_out);
}
